// PINN_Burger2D_85478439125430
// MI455X (gfx1250) — matrix operand order repaired and checked by reading, not compiled, not run on hardware
//
#include <hip/hip_runtime.h>

// CDNA5 / gfx1250 PINN Burgers-2D residual.
// Forward Taylor propagation of 6 channels (v, d_t, d_x, d_y, d_xx, d_yy)
// through the MLP; hidden-layer matvecs done with v_wmma_f32_16x16x32_f16
// (f16 operands, f32 accumulate) since the work is compute-bound
// (~155 GFLOP vs ~2 MB of HBM traffic; weights are L2-resident).

typedef __attribute__((ext_vector_type(16))) _Float16 v16h;
typedef __attribute__((ext_vector_type(8)))  _Float16 v8h;
typedef __attribute__((ext_vector_type(8)))  float    v8f;

union HFrag { v16h v; v8h h[2]; };

#define LAYER_HALF_STRIDE 65536   // 256*256 packed f16 elements per hidden layer
#define NPB 16                    // points per block
#define NU_F 0.0031830988618379067f  // 0.01 / pi

// ---------------------------------------------------------------------------
// Prologue: repack hidden-layer weights (f32 row-major [K][N]) into f16
// fragment-major order for the WMMA B operand (contiguous k, same order as the A reads):
//   fragment = (tile,kc): 32 lanes x 16 halves.
//   lane<16 : N = tile*16 + lane,      K = kc*32 + i        (i = 0..15)
//   lane>=16: N = tile*16 + (lane-16), K = kc*32 + 16 + i
// Each lane then loads its 32 contiguous bytes with two global_load_b128.
// ---------------------------------------------------------------------------
__global__ __launch_bounds__(256) void pack_weights_kernel(
    const float* __restrict__ W1, const float* __restrict__ W2,
    const float* __restrict__ W3, _Float16* __restrict__ packed)
{
  int idx = blockIdx.x * blockDim.x + threadIdx.x;   // 0 .. 3*65536-1
  int layer = idx >> 16;
  int r = idx & 65535;
  const float* W = (layer == 0) ? W1 : ((layer == 1) ? W2 : W3);
  int frag = r >> 9;                 // 0..127  (tile*8 + kc)
  int e    = r & 511;                // element within fragment
  int lane = e >> 4;                 // 0..31
  int i    = e & 15;                 // half index within lane
  int tile = frag >> 3;              // 0..15 -> 16 output neurons each
  int kc   = frag & 7;               // 0..7  -> K chunk of 32
  int nIdx = tile * 16 + (lane & 15);
  int k    = kc * 32 + ((lane >= 16) ? 16 : 0) + i;
  packed[idx] = (_Float16)W[k * 256 + nIdx];
}

// ---------------------------------------------------------------------------
// Main kernel: 256 threads = 8 waves (wave32), 16 points per block.
// ---------------------------------------------------------------------------
__global__ __launch_bounds__(256) void pinn_burgers2d_kernel(
    const float* __restrict__ tin, const float* __restrict__ xin,
    const float* __restrict__ yin,
    const float* __restrict__ W0, const float* __restrict__ b0,
    const float* __restrict__ b1, const float* __restrict__ b2,
    const float* __restrict__ b3,
    const float* __restrict__ W4, const float* __restrict__ b4,
    const _Float16* __restrict__ packed,
    float* __restrict__ out, int n)
{
  // channels: 0=value 1=d_t 2=d_x 3=d_y 4=d_xx 5=d_yy
  __shared__ __align__(16) _Float16 act[6][NPB][256];  // 48 KB
  __shared__ float pts[3][NPB];

  const int tid  = threadIdx.x;
  const int lane = tid & 31;
  const int wave = tid >> 5;
  const int p0   = blockIdx.x * NPB;

  if (tid < 3 * NPB) {
    int c = tid / NPB, m = tid % NPB;
    const float* src = (c == 0) ? tin : ((c == 1) ? xin : yin);
    pts[c][m] = src[p0 + m];
  }
  __syncthreads();

  // ---- Layer 0: 3 -> 256 (K tiny; VALU). Input tangents are the identity:
  //      z_t = W0[0][n], z_x = W0[1][n], z_y = W0[2][n], z_xx = z_yy = 0.
  {
    const int nIdx = tid;
    const float wt = W0[nIdx], wx = W0[256 + nIdx], wy = W0[512 + nIdx];
    const float bb = b0[nIdx];
    #pragma unroll
    for (int m = 0; m < NPB; ++m) {
      float z = fmaf(pts[0][m], wt, fmaf(pts[1][m], wx, fmaf(pts[2][m], wy, bb)));
      float a = tanhf(z);
      float s = 1.0f - a * a;
      float at = s * wt, ax = s * wx, ay = s * wy;
      float axx = -2.0f * a * wx * ax;   // z_xx == 0
      float ayy = -2.0f * a * wy * ay;   // z_yy == 0
      act[0][m][nIdx] = (_Float16)a;
      act[1][m][nIdx] = (_Float16)at;
      act[2][m][nIdx] = (_Float16)ax;
      act[3][m][nIdx] = (_Float16)ay;
      act[4][m][nIdx] = (_Float16)axx;
      act[5][m][nIdx] = (_Float16)ayy;
    }
  }
  __syncthreads();

  // ---- Hidden layers 1..3: 256 -> 256, WMMA.
  for (int layer = 0; layer < 3; ++layer) {
    const _Float16* Wp = packed + layer * LAYER_HALF_STRIDE;
    const float* bias = (layer == 0) ? b1 : ((layer == 1) ? b2 : b3);
    if (layer < 2) __builtin_prefetch(Wp + LAYER_HALF_STRIDE, 0, 1);

    v8f acc[2][6];   // [neuron tile][channel], each 16x16 f32 C tile
    #pragma unroll
    for (int ti = 0; ti < 2; ++ti) {
      const int tile = wave * 2 + ti;
      const float bn = bias[tile * 16 + (lane & 15)];  // bias depends on N only
      #pragma unroll
      for (int ch = 0; ch < 6; ++ch) {
        v8f z = {};
        acc[ti][ch] = z;
      }
      #pragma unroll
      for (int r = 0; r < 8; ++r) acc[ti][0][r] = bn;  // bias -> value channel

      const int m  = lane & 15;          // A row
      const int ko = (lane >> 4) << 4;   // +16 halves for upper half-wave
      for (int kc = 0; kc < 8; ++kc) {
        // B fragment: two coalesced global_load_b128 from packed weights (L2)
        HFrag B;
        const v8h* bp =
            reinterpret_cast<const v8h*>(Wp + (tile * 8 + kc) * 512 + lane * 16);
        B.h[0] = bp[0];
        B.h[1] = bp[1];
        const int kb = kc * 32 + ko;
        #pragma unroll
        for (int ch = 0; ch < 6; ++ch) {
          // A fragment from LDS: halves {kb..kb+15}
          HFrag A;
          A.h[0] = *reinterpret_cast<const v8h*>(&act[ch][m][kb]);
          A.h[1] = *reinterpret_cast<const v8h*>(&act[ch][m][kb + 8]);
          acc[ti][ch] = __builtin_amdgcn_wmma_f32_16x16x32_f16(
              false, A.v, false, B.v, (short)0, acc[ti][ch], false, false);
        }
      }
    }
    __syncthreads();   // all waves done *reading* this layer's activations

    // Elementwise tanh + derivative propagation on the known C layout:
    // element r of v8f <-> M = r + 8*(lane>=16), N = lane&15.
    #pragma unroll
    for (int ti = 0; ti < 2; ++ti) {
      const int ncol  = (wave * 2 + ti) * 16 + (lane & 15);
      const int mBase = (lane >> 4) * 8;
      #pragma unroll
      for (int r = 0; r < 8; ++r) {
        const int m = mBase + r;
        const float zv  = acc[ti][0][r];
        const float zt  = acc[ti][1][r];
        const float zx  = acc[ti][2][r];
        const float zy  = acc[ti][3][r];
        const float zxx = acc[ti][4][r];
        const float zyy = acc[ti][5][r];
        float a = tanhf(zv);
        float s = 1.0f - a * a;
        float at = s * zt, ax = s * zx, ay = s * zy;
        float axx = fmaf(-2.0f * a * zx, ax, s * zxx);
        float ayy = fmaf(-2.0f * a * zy, ay, s * zyy);
        act[0][m][ncol] = (_Float16)a;
        act[1][m][ncol] = (_Float16)at;
        act[2][m][ncol] = (_Float16)ax;
        act[3][m][ncol] = (_Float16)ay;
        act[4][m][ncol] = (_Float16)axx;
        act[5][m][ncol] = (_Float16)ayy;
      }
    }
    __syncthreads();
  }

  // ---- Output layer 256 -> 1: each half-wave reduces one point.
  {
    const int m  = wave * 2 + (lane >> 4);   // point for this half-wave
    const int k0 = lane & 15;
    float s0 = 0.f, s1 = 0.f, s2 = 0.f, s3 = 0.f, s4 = 0.f, s5 = 0.f;
    #pragma unroll 4
    for (int kk = 0; kk < 16; ++kk) {
      const int k = k0 + kk * 16;
      const float wk = W4[k];
      s0 = fmaf(wk, (float)act[0][m][k], s0);
      s1 = fmaf(wk, (float)act[1][m][k], s1);
      s2 = fmaf(wk, (float)act[2][m][k], s2);
      s3 = fmaf(wk, (float)act[3][m][k], s3);
      s4 = fmaf(wk, (float)act[4][m][k], s4);
      s5 = fmaf(wk, (float)act[5][m][k], s5);
    }
    #pragma unroll
    for (int off = 8; off >= 1; off >>= 1) {  // reduce within 16-lane group
      s0 += __shfl_xor(s0, off, 32);
      s1 += __shfl_xor(s1, off, 32);
      s2 += __shfl_xor(s2, off, 32);
      s3 += __shfl_xor(s3, off, 32);
      s4 += __shfl_xor(s4, off, 32);
      s5 += __shfl_xor(s5, off, 32);
    }
    if (k0 == 0) {
      const float u = s0 + b4[0];
      const float f = s1 + u * s2 + u * s3 - NU_F * (s4 + s5);
      out[p0 + m]     = u;   // u output
      out[n + p0 + m] = f;   // f output (concatenated after u)
    }
  }
}

// ---------------------------------------------------------------------------
extern "C" void kernel_launch(void* const* d_in, const int* in_sizes, int n_in,
                              void* d_out, int out_size, void* d_ws, size_t ws_size,
                              hipStream_t stream)
{
  (void)n_in; (void)out_size; (void)ws_size;
  // setup_inputs order: t,x,y, W0,b0, W1,b1, W2,b2, W3,b3, W4,b4 (all f32)
  const float* t  = (const float*)d_in[0];
  const float* x  = (const float*)d_in[1];
  const float* y  = (const float*)d_in[2];
  const float* W0 = (const float*)d_in[3];
  const float* b0 = (const float*)d_in[4];
  const float* W1 = (const float*)d_in[5];
  const float* b1 = (const float*)d_in[6];
  const float* W2 = (const float*)d_in[7];
  const float* b2 = (const float*)d_in[8];
  const float* W3 = (const float*)d_in[9];
  const float* b3 = (const float*)d_in[10];
  const float* W4 = (const float*)d_in[11];
  const float* b4 = (const float*)d_in[12];

  const int n = in_sizes[0];               // 65536 points
  _Float16* packed = (_Float16*)d_ws;      // needs 3*256*256*2 = 384 KB scratch

  // Repack hidden weights to f16 fragment-major every call (deterministic,
  // graph-capture safe, ordered on `stream`).
  pack_weights_kernel<<<(3 * 65536) / 256, 256, 0, stream>>>(W1, W2, W3, packed);

  pinn_burgers2d_kernel<<<n / NPB, 256, 0, stream>>>(
      t, x, y, W0, b0, b1, b2, b3, W4, b4, packed, (float*)d_out, n);
}
